// InstructionNet_76871324664282
// MI455X (gfx1250) — matrix operand order repaired and checked by reading, not compiled, not run on hardware
//
#include <hip/hip_runtime.h>
#include <math.h>

// ---------------- model constants ----------------
#define NTOK   8192      // B*L
#define SEQL   2048
#define DMODEL 512
#define DKH    64
#define DFFN   1365
#define DFFP   1376      // DFFN padded to 16-element multiple (16B-aligned bf16 rows)
#define WINSZ  128
#define NBH    32        // B*H
#define NATT   512       // NBH * (SEQL/WINSZ) attention blocks

typedef __attribute__((ext_vector_type(16))) __bf16 v16bf;
typedef __attribute__((ext_vector_type(8)))  float  v8f;

union Frag {
    v16bf v;
    uint4 q[2];
    unsigned short h[16];
};

__device__ __forceinline__ unsigned short f2bf(float f) {
    unsigned int u = __float_as_uint(f);
    u += 0x7FFFu + ((u >> 16) & 1u);       // round-to-nearest-even
    return (unsigned short)(u >> 16);
}

__device__ __forceinline__ float siluf(float x) { return x / (1.0f + expf(-x)); }
__device__ __forceinline__ float softplusf(float x) {
    return (x > 20.0f) ? x : log1pf(expf(x));
}

#define WMMA_BF16(A, B, C) \
    __builtin_amdgcn_wmma_f32_16x16x32_bf16(false, (A), false, (B), (short)0, (C), false, false)

// ---------------------------------------------------------------------------
// Weight convert + transpose: f32 [M,K] row-major -> bf16 Wt[Kp][ldM] K-major,
// zero-padded in both dimensions (so GEMM tile reads need no guards).
// ---------------------------------------------------------------------------
__global__ __launch_bounds__(256) void cvt_wt_kernel(const float* __restrict__ src,
                                                     unsigned short* __restrict__ dst,
                                                     int M, int K, int ldM, int Kp)
{
    int i = blockIdx.x * 256 + threadIdx.x;
    if (i >= Kp * ldM) return;
    int k = i / ldM, m = i % ldM;
    dst[i] = (k < K && m < M) ? f2bf(src[(size_t)m * K + k]) : (unsigned short)0;
}

// ---------------------------------------------------------------------------
// Generic WMMA GEMM (bf16 x bf16 -> f32):
//   C[n, colOff+m] = A[n,:K] @ W[m,:K]^T (+bias[m]) (+resid)
// A:  bf16 [NTOK, ldA] (rows 16B-aligned, cols K..(32-mult) zero-padded).
// Wt: bf16 [Kp, ldM]  K-major (pre-transposed, zero-padded).
// Block: 256 threads (8 waves), tile 128 rows x 64 cols, K stepped by 32.
// Fully unguarded b128 staging; 4 v_wmma per wave per K-step.
// ---------------------------------------------------------------------------
__global__ __launch_bounds__(256) void gemm_kernel(
    const unsigned short* __restrict__ A, int ldA,
    const unsigned short* __restrict__ Wt, int ldM,
    const float* __restrict__ bias,
    const float* __restrict__ resid,
    float* __restrict__ C, int ldC, int colOff,
    int K, int M)
{
    __shared__ __align__(16) unsigned short As[128 * 32];
    __shared__ __align__(16) unsigned short Bs[32 * 64];   // Bs[k][n]

    const int tid  = threadIdx.x;
    const int lane = tid & 31, wave = tid >> 5;
    const int rowBase = blockIdx.y * 128;
    const int colBase = blockIdx.x * 64;
    const int half8 = (lane >> 4) * 8, mrow = lane & 15;

    // staging assignments (uniform, zero divergence in the K loop)
    const int ar = tid >> 1, ac = (tid & 1) * 16;   // A: 128x32, 2x uint4/thread
    const int bk = tid >> 3, bn = (tid & 7) * 8;    // B: 32x64,  1x uint4/thread
    const size_t Arow  = (size_t)(rowBase + ar) * ldA + ac;
    const size_t Wbase = (size_t)bk * ldM + colBase + bn;

    v8f z = {0.f,0.f,0.f,0.f,0.f,0.f,0.f,0.f};
    v8f acc0 = z, acc1 = z, acc2 = z, acc3 = z;

    const int ksteps = (K + 31) / 32;
    for (int kt = 0; kt < ksteps; ++kt) {
        const int k0 = kt * 32;
        {   // A tile: straight b128 copy global->LDS
            const uint4* src = (const uint4*)&A[Arow + k0];
            uint4* dst = (uint4*)&As[ar * 32 + ac];
            dst[0] = src[0];
            dst[1] = src[1];
        }
        {   // B tile: straight b128 copy global->LDS (already K-major)
            *(uint4*)&Bs[bk * 64 + bn] = *(const uint4*)&Wt[Wbase + (size_t)k0 * ldM];
        }
        if (kt + 1 < ksteps)
            __builtin_prefetch(&Wt[Wbase + (size_t)(k0 + 32) * ldM], 0, 1);
        __syncthreads();

        Frag a;
        const unsigned short* ap = &As[(wave * 16 + mrow) * 32];
        a.q[0] = *(const uint4*)(ap + half8);
        a.q[1] = *(const uint4*)(ap + 16 + half8);

        Frag b;
        const unsigned short* bp = &Bs[mrow];
#pragma unroll
        for (int s = 0; s < 16; ++s) b.h[s] = bp[((s >> 3) * 16 + half8 + (s & 7)) * 64 +  0];
        acc0 = WMMA_BF16(a.v, b.v, acc0);
#pragma unroll
        for (int s = 0; s < 16; ++s) b.h[s] = bp[((s >> 3) * 16 + half8 + (s & 7)) * 64 + 16];
        acc1 = WMMA_BF16(a.v, b.v, acc1);
#pragma unroll
        for (int s = 0; s < 16; ++s) b.h[s] = bp[((s >> 3) * 16 + half8 + (s & 7)) * 64 + 32];
        acc2 = WMMA_BF16(a.v, b.v, acc2);
#pragma unroll
        for (int s = 0; s < 16; ++s) b.h[s] = bp[((s >> 3) * 16 + half8 + (s & 7)) * 64 + 48];
        acc3 = WMMA_BF16(a.v, b.v, acc3);
        __syncthreads();
    }

    const int nloc = lane & 15, mbase = (lane >> 4) * 8;
    const int grow0 = rowBase + wave * 16 + mbase;
    v8f accs[4] = {acc0, acc1, acc2, acc3};
#pragma unroll
    for (int ni = 0; ni < 4; ++ni) {
        int gcol = colBase + ni * 16 + nloc;
        if (gcol < M) {
            float bv = bias ? bias[gcol] : 0.f;
#pragma unroll
            for (int r = 0; r < 8; ++r) {
                size_t idx = (size_t)(grow0 + r) * ldC + colOff + gcol;
                float v = accs[ni][r] + bv;
                if (resid) v += resid[idx];
                C[idx] = v;
            }
        }
    }
}

// ---------------------------------------------------------------------------
// Encoder prep
// ---------------------------------------------------------------------------
__global__ __launch_bounds__(256) void feat_kernel(const int* __restrict__ x,
                                                   unsigned short* __restrict__ featb)
{
    int i = blockIdx.x * 256 + threadIdx.x;        // NTOK*288
    int t = i / 288, c = i % 288;
    featb[i] = f2bf((float)x[(size_t)t * 289 + 1 + c]);
}

__global__ __launch_bounds__(256) void te_kernel(const int* __restrict__ x,
                                                 const float* __restrict__ emb,
                                                 float* __restrict__ cat)
{
    int i = blockIdx.x * 256 + threadIdx.x;        // NTOK*192
    int t = i / 192, c = i % 192;
    int tok = x[(size_t)t * 289];
    cat[(size_t)t * 768 + c] = emb[(size_t)tok * 192 + c];
}

__global__ __launch_bounds__(256) void silu_cvt_kernel(const float* __restrict__ in,
                                                       unsigned short* __restrict__ out, int n)
{
    int i = blockIdx.x * 256 + threadIdx.x;
    if (i < n) out[i] = f2bf(siluf(in[i]));
}

// f = silu(u) * g, written as bf16 with row padding DFFN -> DFFP (pad = 0)
__global__ __launch_bounds__(256) void silu_mul_pad_kernel(const float* __restrict__ u,
                                                           const float* __restrict__ g,
                                                           unsigned short* __restrict__ ub)
{
    int i = blockIdx.x * 256 + threadIdx.x;        // NTOK*DFFP
    int r = i / DFFP, c = i % DFFP;
    float v = 0.f;
    if (c < DFFN) {
        size_t s = (size_t)r * DFFN + c;
        v = siluf(u[s]) * g[s];
    }
    ub[i] = f2bf(v);
}

// ---------------------------------------------------------------------------
// Row-wise norms: one wave32 per row, 8 rows per 256-thread block.
// ---------------------------------------------------------------------------
__global__ __launch_bounds__(256) void rms_kernel(const float* __restrict__ x,
                                                  const float* __restrict__ w,
                                                  unsigned short* __restrict__ out, int Dm)
{
    int wave = threadIdx.x >> 5, lane = threadIdx.x & 31;
    size_t row = (size_t)blockIdx.x * 8 + wave;
    const float* xr = x + row * Dm;
    float ss = 0.f;
    for (int c = lane; c < Dm; c += 32) { float v = xr[c]; ss += v * v; }
    for (int o = 16; o > 0; o >>= 1) ss += __shfl_xor(ss, o, 32);
    float sc = rsqrtf(ss / (float)Dm + 1.1920929e-07f);
    unsigned short* orow = out + row * Dm;
    for (int c = lane; c < Dm; c += 32) orow[c] = f2bf(xr[c] * sc * w[c]);
}

__global__ __launch_bounds__(256) void ln_silu_kernel(const float* __restrict__ x,
                                                      const float* __restrict__ w,
                                                      const float* __restrict__ b,
                                                      float* __restrict__ outf,
                                                      unsigned short* __restrict__ outb, int Dm)
{
    int wave = threadIdx.x >> 5, lane = threadIdx.x & 31;
    size_t row = (size_t)blockIdx.x * 8 + wave;
    const float* xr = x + row * Dm;
    float s = 0.f;
    for (int c = lane; c < Dm; c += 32) s += xr[c];
    for (int o = 16; o > 0; o >>= 1) s += __shfl_xor(s, o, 32);
    float m = s / (float)Dm;
    float vs = 0.f;
    for (int c = lane; c < Dm; c += 32) { float d = xr[c] - m; vs += d * d; }
    for (int o = 16; o > 0; o >>= 1) vs += __shfl_xor(vs, o, 32);
    float inv = rsqrtf(vs / (float)Dm + 1e-5f);
    for (int c = lane; c < Dm; c += 32) {
        float y = (xr[c] - m) * inv * w[c] + b[c];
        float sv = siluf(y);
        if (outf) outf[row * Dm + c] = sv;
        if (outb) outb[row * Dm + c] = f2bf(sv);
    }
}

// ---------------------------------------------------------------------------
// Rope: one thread per (b,h,pos). qkv channel layout: h*192 + d*3 + {q,k,v}.
// ---------------------------------------------------------------------------
__global__ __launch_bounds__(256) void rope_kernel(const float* __restrict__ qkv,
                                                   unsigned short* __restrict__ qb,
                                                   unsigned short* __restrict__ kb,
                                                   unsigned short* __restrict__ vb)
{
    int idx = blockIdx.x * 256 + threadIdx.x;      // bh*SEQL + pos
    int pos = idx & (SEQL - 1);
    int bh  = idx >> 11;
    int b   = bh >> 3, h = bh & 7;
    const float* src = qkv + ((size_t)b * SEQL + pos) * (3 * DMODEL) + h * (DKH * 3);
    size_t dst = (size_t)idx * DKH;
    for (int p = 0; p < 32; ++p) {
        float inv = expf(-((2.0f * (float)p) / 64.0f) * 9.210340371976184f); // 10000^(-2p/64)
        float ang = (float)pos * inv;
        float c = cosf(ang), s = sinf(ang);
        float q1 = src[(2 * p) * 3 + 0], q2 = src[(2 * p + 1) * 3 + 0];
        float k1 = src[(2 * p) * 3 + 1], k2 = src[(2 * p + 1) * 3 + 1];
        qb[dst + 2 * p]     = f2bf(q1 * c - q2 * s);
        qb[dst + 2 * p + 1] = f2bf(q1 * s + q2 * c);
        kb[dst + 2 * p]     = f2bf(k1 * c - k2 * s);
        kb[dst + 2 * p + 1] = f2bf(k1 * s + k2 * c);
        vb[dst + 2 * p]     = f2bf(src[(2 * p) * 3 + 2]);
        vb[dst + 2 * p + 1] = f2bf(src[(2 * p + 1) * 3 + 2]);
    }
}

// ---------------------------------------------------------------------------
// Attention scores: one workgroup per (b,h,blk).  S[128x256] = Q @ Kc^T * 0.125
// K staged *transposed* in LDS; each K^T fragment is 16 strided 16-bit reads.
// ---------------------------------------------------------------------------
__global__ __launch_bounds__(256) void attn_scores_kernel(
    const unsigned short* __restrict__ qb,
    const unsigned short* __restrict__ kb,
    float* __restrict__ S)
{
    __shared__ __align__(16) unsigned short Qs[128 * 64];
    __shared__ __align__(16) unsigned short Kt[64 * 256];  // Kt[d][j]

    const int blkId = blockIdx.x;
    const int blk = blkId & 15, bh = blkId >> 4;
    const int tid = threadIdx.x;

    {   // Q stage: vectorized copy
        const uint4* src = (const uint4*)(qb + ((size_t)bh * SEQL + (size_t)blk * WINSZ) * DKH);
        uint4* dst = (uint4*)Qs;
        for (int i = tid; i < 1024; i += 256) dst[i] = src[i];
    }
    {   // K stage: one key row per thread, transposed scatter
        const int j = tid;
        const long g = ((long)blk * WINSZ - WINSZ + j) * DKH;
        if (g >= 0) {
            const uint4* src = (const uint4*)&kb[(size_t)bh * SEQL * DKH + (size_t)g];
#pragma unroll
            for (int c4 = 0; c4 < 8; ++c4) {
                uint4 v = src[c4];
                const unsigned short* vs = (const unsigned short*)&v;
#pragma unroll
                for (int e = 0; e < 8; ++e) Kt[(c4 * 8 + e) * 256 + j] = vs[e];
            }
        } else {
#pragma unroll
            for (int d = 0; d < 64; ++d) Kt[d * 256 + j] = 0;
        }
    }
    __syncthreads();

    const int lane = tid & 31, wave = tid >> 5;
    const int half8 = (lane >> 4) * 8, mrow = lane & 15;

    Frag a0, a1;
    const unsigned short* ap = &Qs[(wave * 16 + mrow) * 64];
    a0.q[0] = *(const uint4*)(ap + half8);
    a0.q[1] = *(const uint4*)(ap + 16 + half8);
    a1.q[0] = *(const uint4*)(ap + 32 + half8);
    a1.q[1] = *(const uint4*)(ap + 48 + half8);

    float* sblk = S + (size_t)blkId * 128 * 256;
    const int jn = lane & 15, mbase = (lane >> 4) * 8;

    for (int ni = 0; ni < 16; ++ni) {
        v8f acc = {0.f,0.f,0.f,0.f,0.f,0.f,0.f,0.f};
        Frag b;
        const unsigned short* bp0 = &Kt[ni * 16 + mrow];
#pragma unroll
        for (int s = 0; s < 16; ++s) b.h[s] = bp0[((s >> 3) * 16 + half8 + (s & 7)) * 256];
        acc = WMMA_BF16(a0.v, b.v, acc);
        const unsigned short* bp1 = &Kt[32 * 256 + ni * 16 + mrow];
#pragma unroll
        for (int s = 0; s < 16; ++s) b.h[s] = bp1[((s >> 3) * 16 + half8 + (s & 7)) * 256];
        acc = WMMA_BF16(a1.v, b.v, acc);

        const int j = ni * 16 + jn;
#pragma unroll
        for (int r = 0; r < 8; ++r) {
            int i = wave * 16 + mbase + r;
            bool valid = (j >= i) && (j <= i + WINSZ) && (blk > 0 || j >= WINSZ);
            sblk[(size_t)i * 256 + j] = valid ? acc[r] * 0.125f : -1e30f;
        }
    }
}

// Row softmax: one thread per 256-score row -> bf16 P.
__global__ __launch_bounds__(256) void softmax_kernel(const float* __restrict__ S,
                                                      unsigned short* __restrict__ P)
{
    size_t row = (size_t)blockIdx.x * 256 + threadIdx.x;
    const float* s = S + row * 256;
    unsigned short* p = P + row * 256;
    float mx = -3.4e38f;
    for (int j = 0; j < 256; ++j) mx = fmaxf(mx, s[j]);
    float sum = 0.f;
    for (int j = 0; j < 256; ++j) sum += expf(s[j] - mx);
    float inv = 1.0f / sum;
    for (int j = 0; j < 256; ++j) p[j] = f2bf(expf(s[j] - mx) * inv);
}

// ---------------------------------------------------------------------------
// O = P[128x256] @ V[256x64]; output written as bf16 in [token, h*64+d] layout
// so the out-projection GEMM consumes it directly.
// ---------------------------------------------------------------------------
__global__ __launch_bounds__(256) void attn_o_kernel(
    const unsigned short* __restrict__ P,
    const unsigned short* __restrict__ vb,
    unsigned short* __restrict__ Ob)
{
    __shared__ __align__(16) unsigned short Vs[256 * 64];

    const int blkId = blockIdx.x;
    const int blk = blkId & 15, bh = blkId >> 4;
    const int b = bh >> 3, h = bh & 7;
    const int tid = threadIdx.x;

    {   // V stage: one row per thread, vectorized
        const int j = tid;
        const long g = ((long)blk * WINSZ - WINSZ + j) * DKH;
        uint4* dst = (uint4*)&Vs[j * 64];
        if (g >= 0) {
            const uint4* src = (const uint4*)&vb[(size_t)bh * SEQL * DKH + (size_t)g];
#pragma unroll
            for (int c4 = 0; c4 < 8; ++c4) dst[c4] = src[c4];
        } else {
            uint4 zz = {0u, 0u, 0u, 0u};
#pragma unroll
            for (int c4 = 0; c4 < 8; ++c4) dst[c4] = zz;
        }
    }
    __syncthreads();

    const int lane = tid & 31, wave = tid >> 5;
    const int half8 = (lane >> 4) * 8, mrow = lane & 15;

    v8f z = {0.f,0.f,0.f,0.f,0.f,0.f,0.f,0.f};
    v8f acc0 = z, acc1 = z, acc2 = z, acc3 = z;

    const unsigned short* prow = P + ((size_t)blkId * 128 + wave * 16 + mrow) * 256;
#pragma unroll
    for (int kc = 0; kc < 8; ++kc) {
        Frag a;
        a.q[0] = *(const uint4*)(prow + kc * 32 + half8);
        a.q[1] = *(const uint4*)(prow + kc * 32 + 16 + half8);
        const unsigned short* vrow = &Vs[(kc * 32) * 64 + mrow];
        Frag bf;
#pragma unroll
        for (int s = 0; s < 16; ++s) bf.h[s] = vrow[((s >> 3) * 16 + half8 + (s & 7)) * 64 +  0];
        acc0 = WMMA_BF16(a.v, bf.v, acc0);
#pragma unroll
        for (int s = 0; s < 16; ++s) bf.h[s] = vrow[((s >> 3) * 16 + half8 + (s & 7)) * 64 + 16];
        acc1 = WMMA_BF16(a.v, bf.v, acc1);
#pragma unroll
        for (int s = 0; s < 16; ++s) bf.h[s] = vrow[((s >> 3) * 16 + half8 + (s & 7)) * 64 + 32];
        acc2 = WMMA_BF16(a.v, bf.v, acc2);
#pragma unroll
        for (int s = 0; s < 16; ++s) bf.h[s] = vrow[((s >> 3) * 16 + half8 + (s & 7)) * 64 + 48];
        acc3 = WMMA_BF16(a.v, bf.v, acc3);
    }

    const int nloc = lane & 15, mbase = (lane >> 4) * 8;
#pragma unroll
    for (int r = 0; r < 8; ++r) {
        int i = wave * 16 + mbase + r;
        int pos = blk * WINSZ + i;
        unsigned short* orow = Ob + ((size_t)b * SEQL + pos) * DMODEL + h * DKH;
        orow[ 0 + nloc] = f2bf(acc0[r]);
        orow[16 + nloc] = f2bf(acc1[r]);
        orow[32 + nloc] = f2bf(acc2[r]);
        orow[48 + nloc] = f2bf(acc3[r]);
    }
}

// ---------------------------------------------------------------------------
// Head post-processing: one thread per token, y[31] -> out[40].
// ---------------------------------------------------------------------------
__global__ __launch_bounds__(256) void head_final_kernel(const float* __restrict__ y,
                                                         float* __restrict__ out)
{
    int t = blockIdx.x * 256 + threadIdx.x;
    const float* yr = y + (size_t)t * 31;
    float* o = out + (size_t)t * 40;

    int fp = 0; float fb = yr[0];
    for (int j = 1; j < 11; ++j) if (yr[j] > fb) { fb = yr[j]; fp = j; }
    float fc_reg = softplusf(yr[11]);
    float fetch = (fp < 10) ? (float)(fp + 1) : fc_reg;

    int ep = 0; float eb = yr[12];
    for (int j = 1; j < 11; ++j) if (yr[12 + j] > eb) { eb = yr[12 + j]; ep = j; }
    float ec_reg = softplusf(yr[23]);
    float execc = (ep < 10) ? (float)(ep + 1) : ec_reg;

    float bm_log = yr[24];
    float bm = 1.0f / (1.0f + expf(-bm_log));

    float icm = fmaxf(yr[25], fmaxf(yr[26], yr[27]));
    float e0 = expf(yr[25] - icm), e1 = expf(yr[26] - icm), e2 = expf(yr[27] - icm);
    float es = e0 + e1 + e2;
    float dcm = fmaxf(yr[28], fmaxf(yr[29], yr[30]));
    float d0 = expf(yr[28] - dcm), d1 = expf(yr[29] - dcm), d2 = expf(yr[30] - dcm);
    float ds = d0 + d1 + d2;

    o[0] = fetch;
    for (int j = 0; j < 11; ++j) o[1 + j] = yr[j];
    o[12] = fc_reg;
    o[13] = execc;
    for (int j = 0; j < 11; ++j) o[14 + j] = yr[12 + j];
    o[25] = ec_reg;
    o[26] = bm;
    o[27] = e0 / es; o[28] = e1 / es; o[29] = e2 / es;
    o[30] = d0 / ds; o[31] = d1 / ds; o[32] = d2 / ds;
    o[33] = bm_log;
    o[34] = yr[25]; o[35] = yr[26]; o[36] = yr[27];
    o[37] = yr[28]; o[38] = yr[29]; o[39] = yr[30];
}

// ---------------------------------------------------------------------------
// Host orchestration
// ---------------------------------------------------------------------------
extern "C" void kernel_launch(void* const* d_in, const int* in_sizes, int n_in,
                              void* d_out, int out_size, void* d_ws, size_t ws_size,
                              hipStream_t stream)
{
    (void)in_sizes; (void)n_in; (void)out_size; (void)ws_size;

    const int*   x        = (const int*)  d_in[0];
    const float* emb      = (const float*)d_in[1];
    const float* reg_w    = (const float*)d_in[2];
    const float* reg_b    = (const float*)d_in[3];
    const float* br_w     = (const float*)d_in[4];
    const float* br_b     = (const float*)d_in[5];
    const float* sh_w     = (const float*)d_in[6];
    const float* sh_b     = (const float*)d_in[7];
    const float* inst_w   = (const float*)d_in[8];
    const float* inst_b   = (const float*)d_in[9];
    const float* enc_ln_w = (const float*)d_in[10];
    const float* enc_ln_b = (const float*)d_in[11];
    const float* qkv_w    = (const float*)d_in[12];
    const float* out_w    = (const float*)d_in[13];
    const float* out_b    = (const float*)d_in[14];
    const float* w1_w     = (const float*)d_in[15];
    const float* w1_b     = (const float*)d_in[16];
    const float* w3_w     = (const float*)d_in[17];
    const float* w3_b     = (const float*)d_in[18];
    const float* w2_w     = (const float*)d_in[19];
    const float* w2_b     = (const float*)d_in[20];
    const float* n1_w     = (const float*)d_in[21];
    const float* n2_w     = (const float*)d_in[22];
    const float* hl_w     = (const float*)d_in[23];
    const float* hl_b     = (const float*)d_in[24];
    const float* head_w   = (const float*)d_in[25];
    const float* head_b   = (const float*)d_in[26];
    float* outp = (float*)d_out;

    // ---- workspace arena ----
    char* base = (char*)d_ws;
    size_t off = 0;
    auto alloc = [&](size_t bytes) {
        size_t o = off; off += (bytes + 255) & ~(size_t)255; return o;
    };
    const size_t MB = 1024 * 1024;
    typedef unsigned short ush;

    float* h   = (float*)(base + alloc((size_t)NTOK * DMODEL * 4));
    float* t   = (float*)(base + alloc((size_t)NTOK * DMODEL * 4));
    float* qkv = (float*)(base + alloc((size_t)NTOK * 3 * DMODEL * 4));
    float* y   = (float*)(base + alloc((size_t)NTOK * 31 * 4));
    ush* tb    = (ush*)(base + alloc((size_t)NTOK * DMODEL * 2));
    ush* qb    = (ush*)(base + alloc((size_t)NBH * SEQL * DKH * 2));
    ush* kb    = (ush*)(base + alloc((size_t)NBH * SEQL * DKH * 2));
    ush* vbuf  = (ush*)(base + alloc((size_t)NBH * SEQL * DKH * 2));
    ush* Ob    = (ush*)(base + alloc((size_t)NTOK * DMODEL * 2));
    ush* featb = (ush*)(base + alloc((size_t)NTOK * 288 * 2));
    ush* catb  = (ush*)(base + alloc((size_t)NTOK * 768 * 2));
    ush* ub    = (ush*)(base + alloc((size_t)NTOK * DFFP * 2));
    // bf16 K-major weight arena (re-built every call; deterministic, ~21 MB)
    ush* regb  = (ush*)(base + alloc((size_t)64  * 192  * 2));   // [Kp=64][192]
    ush* brb   = (ush*)(base + alloc((size_t)32  * 192  * 2));   // [32][192]
    ush* shb   = (ush*)(base + alloc((size_t)192 * 192  * 2));   // [192][192]
    ush* instb = (ush*)(base + alloc((size_t)768 * 512  * 2));   // [768][512]
    ush* qkvb  = (ush*)(base + alloc((size_t)3 * 512 * 1536 * 2)); // [512][1536]/layer
    ush* outb  = (ush*)(base + alloc((size_t)3 * 512 * 512  * 2)); // [512][512]/layer
    ush* w1b   = (ush*)(base + alloc((size_t)3 * 512 * 1408 * 2)); // [512][1408]/layer
    ush* w3b   = (ush*)(base + alloc((size_t)3 * 512 * 1408 * 2)); // [512][1408]/layer
    ush* w2b   = (ush*)(base + alloc((size_t)3 * 1376 * 512 * 2)); // [Kp=1376][512]/layer
    ush* headb = (ush*)(base + alloc((size_t)512 * 64  * 2));      // [512][64]
    // shared region (disjoint lifetimes): encoder cat(f32) | attn S/P | FFN u/g
    size_t regionOff = alloc(96 * MB);
    float* S   = (float*)(base + regionOff);                     // 64 MiB
    ush*   P   = (ush*)(base + regionOff + 64 * MB);             // 32 MiB
    float* u   = (float*)(base + regionOff);                     // 44.7 MB
    float* g   = (float*)(base + regionOff + 45 * MB);           // 44.7 MB
    float* cat = (float*)(base + regionOff);                     // 25.2 MB

    auto gemm = [&](const ush* A, int ldA, const ush* Wt, int ldM,
                    const float* bias, const float* resid,
                    float* C, int ldC, int colOff, int K, int M) {
        dim3 grid((M + 63) / 64, NTOK / 128);
        gemm_kernel<<<grid, 256, 0, stream>>>(A, ldA, Wt, ldM, bias, resid,
                                              C, ldC, colOff, K, M);
    };
    auto cvt = [&](const float* src, ush* dst, int M, int K, int ldM, int Kp) {
        int n = Kp * ldM;
        cvt_wt_kernel<<<(n + 255) / 256, 256, 0, stream>>>(src, dst, M, K, ldM, Kp);
    };

    // ---- one-shot weight convert+transpose to K-major bf16 ----
    cvt(reg_w,  regb,  192, 64,  192, 64);
    cvt(br_w,   brb,   192, 32,  192, 32);
    cvt(sh_w,   shb,   192, 192, 192, 192);
    cvt(inst_w, instb, 512, 768, 512, 768);
    for (int l = 0; l < 3; ++l) {
        cvt(qkv_w + (size_t)l * 1536 * 512, qkvb + (size_t)l * 512 * 1536, 1536, 512, 1536, 512);
        cvt(out_w + (size_t)l * 512 * 512,  outb + (size_t)l * 512 * 512,   512, 512,  512, 512);
        cvt(w1_w  + (size_t)l * DFFN * 512, w1b  + (size_t)l * 512 * 1408, DFFN, 512, 1408, 512);
        cvt(w3_w  + (size_t)l * DFFN * 512, w3b  + (size_t)l * 512 * 1408, DFFN, 512, 1408, 512);
        cvt(w2_w  + (size_t)l * 512 * DFFN, w2b  + (size_t)l * 1376 * 512,  512, DFFN, 512, 1376);
    }
    cvt(head_w, headb, 31, 512, 64, 512);

    // ---- encoder ----
    feat_kernel<<<(NTOK * 288) / 256, 256, 0, stream>>>(x, featb);
    te_kernel<<<(NTOK * 192) / 256, 256, 0, stream>>>(x, emb, cat);
    gemm(featb +   0, 288, regb, 192, reg_b, nullptr, cat, 768, 192,  64, 192);
    gemm(featb + 256, 288, brb,  192, br_b,  nullptr, cat, 768, 384,  32, 192);
    gemm(featb +  64, 288, shb,  192, sh_b,  nullptr, cat, 768, 576, 192, 192);
    silu_cvt_kernel<<<(NTOK * 768) / 256, 256, 0, stream>>>(cat, catb, NTOK * 768);
    gemm(catb, 768, instb, 512, inst_b, nullptr, t, DMODEL, 0, 768, DMODEL);
    ln_silu_kernel<<<NTOK / 8, 256, 0, stream>>>(t, enc_ln_w, enc_ln_b, h, nullptr, DMODEL);

    // ---- transformer layers ----
    for (int l = 0; l < 3; ++l) {
        rms_kernel<<<NTOK / 8, 256, 0, stream>>>(h, n1_w + l * DMODEL, tb, DMODEL);
        gemm(tb, DMODEL, qkvb + (size_t)l * 512 * 1536, 1536, nullptr, nullptr,
             qkv, 3 * DMODEL, 0, DMODEL, 3 * DMODEL);
        rope_kernel<<<(NBH * SEQL) / 256, 256, 0, stream>>>(qkv, qb, kb, vbuf);
        attn_scores_kernel<<<NATT, 256, 0, stream>>>(qb, kb, S);
        softmax_kernel<<<(NATT * 128) / 256, 256, 0, stream>>>(S, P);
        attn_o_kernel<<<NATT, 256, 0, stream>>>(P, vbuf, Ob);
        gemm(Ob, DMODEL, outb + (size_t)l * 512 * 512, 512, out_b + l * DMODEL,
             h, h, DMODEL, 0, DMODEL, DMODEL);

        rms_kernel<<<NTOK / 8, 256, 0, stream>>>(h, n2_w + l * DMODEL, tb, DMODEL);
        gemm(tb, DMODEL, w1b + (size_t)l * 512 * 1408, 1408, w1_b + l * DFFN, nullptr,
             u, DFFN, 0, DMODEL, DFFN);
        gemm(tb, DMODEL, w3b + (size_t)l * 512 * 1408, 1408, w3_b + l * DFFN, nullptr,
             g, DFFN, 0, DMODEL, DFFN);
        silu_mul_pad_kernel<<<(NTOK * DFFP) / 256, 256, 0, stream>>>(u, g, ub);
        gemm(ub, DFFP, w2b + (size_t)l * 1376 * 512, 512, w2_b + l * DMODEL,
             h, h, DMODEL, 0, DFFN, DMODEL);
    }

    // ---- head ----
    ln_silu_kernel<<<NTOK / 8, 256, 0, stream>>>(h, hl_w, hl_b, nullptr, tb, DMODEL);
    gemm(tb, DMODEL, headb, 64, head_b, nullptr, y, 31, 0, DMODEL, 31);
    head_final_kernel<<<NTOK / 256, 256, 0, stream>>>(y, outp);
}
